// CausalSelfAttention_67585605370191
// MI455X (gfx1250) — compile-verified, matrix operand order checked by reading, not run on hardware
//
#include <hip/hip_runtime.h>

// ---------------------------------------------------------------------------
// CausalSelfAttention (non-causal softmax path, per reference) on gfx1250.
// All three matmuls (QKV, QK^T/PV, proj) run on v_wmma_f32_16x16x32_bf16.
// K/A tiles staged with GLOBAL_LOAD_ASYNC_TO_LDS_B128 (ASYNCcnt tracked).
// H=1024, NH=16, HD=64, B=2, T=2048  =>  M = B*T = 4096 token rows.
// ---------------------------------------------------------------------------

#define H_ 1024
#define NH_ 16
#define HD_ 64
#define B_ 2
#define T_ 2048
#define M_ (B_ * T_)          // 4096
#define N_QKV (3 * H_)        // 3072

typedef __attribute__((ext_vector_type(16))) __bf16 v16bf;
typedef __attribute__((ext_vector_type(8)))  float  v8f;
typedef __attribute__((ext_vector_type(4)))  int    v4i_;

union FragU {
  uint4 q[2];
  unsigned short h[16];
  v16bf b;
};

__device__ __forceinline__ unsigned short f2bf(float x) {
  unsigned int u = __float_as_uint(x);
  u += 0x7fffu + ((u >> 16) & 1u);   // round-to-nearest-even
  return (unsigned short)(u >> 16);
}

// ---------------------------------------------------------------------------
// Async global->LDS copy (16 bytes per lane). Signature (confirmed by compiler
// diagnostic): (as1 int4* src, as3 int4* dst, imm offset, imm cpol).
// ---------------------------------------------------------------------------
#if __has_builtin(__builtin_amdgcn_global_load_async_to_lds_b128)
#define ASYNC_LDS 1
#else
#define ASYNC_LDS 0
#endif

__device__ __forceinline__ void stage_b128(const unsigned short* gsrc,
                                           unsigned short* ldst) {
#if ASYNC_LDS
  typedef __attribute__((address_space(1))) v4i_* gasp_t;
  typedef __attribute__((address_space(3))) v4i_* lasp_t;
  // generic global pointer value == as(1) value; low 32 bits of a generic
  // shared pointer are the LDS byte offset (ISA aperture mapping).
  __builtin_amdgcn_global_load_async_to_lds_b128(
      (gasp_t)(uintptr_t)gsrc, (lasp_t)(unsigned int)(uintptr_t)ldst, 0, 0);
#else
  *(uint4*)ldst = *(const uint4*)gsrc;
#endif
}

__device__ __forceinline__ void stage_wait() {
#if ASYNC_LDS
#if __has_builtin(__builtin_amdgcn_s_wait_asynccnt)
  __builtin_amdgcn_s_wait_asynccnt(0);
#else
  asm volatile("s_wait_asynccnt 0x0" ::: "memory");
#endif
#endif
}

// Load one 16x32 bf16 WMMA operand fragment.
// For the A operand: p = tile base ([row][k] layout, row stride = ld halves).
// For the B operand: p = base of a *transposed* [n][k] tile; lane indexes n.
// Per ISA: lanes 0-15 hold K {0..7,16..23}, lanes 16-31 hold K {8..15,24..31}.
__device__ __forceinline__ v16bf ld_frag(const unsigned short* p, int ld, int koff) {
  int lane = threadIdx.x & 31;
  int r    = lane & 15;
  int hi   = (lane >> 4) << 3;       // 0 or 8
  const unsigned short* s0 = p + r * ld + koff + hi;
  FragU f;
  f.q[0] = *(const uint4*)(s0);
  f.q[1] = *(const uint4*)(s0 + 16);
  return f.b;
}

__device__ __forceinline__ v8f wmma_bf16(v16bf a, v16bf b, v8f c) {
  return __builtin_amdgcn_wmma_f32_16x16x32_bf16(
      /*neg_a=*/false, a, /*neg_b=*/false, b,
      /*c_mod=*/(short)0, c, /*reuse_a=*/false, /*reuse_b=*/false);
}

// ---------------------------------------------------------------------------
// f32 -> bf16 conversion
// ---------------------------------------------------------------------------
__global__ void cvt_f32_bf16(const float* __restrict__ in,
                             unsigned short* __restrict__ out, int n) {
  int i = blockIdx.x * blockDim.x + threadIdx.x;
  int stride = gridDim.x * blockDim.x;
  for (; i < n; i += stride) out[i] = f2bf(in[i]);
}

// ---------------------------------------------------------------------------
// Tiled bf16 GEMM:  C[M,N] = A[M,K] * B[K,N] + bias[N]
// Block tile 64(M) x 128(N), 256 threads = 8 waves, wave tile 32x32 (2x2 WMMA).
// MODE 0: scatter into q/k/v bf16 [b,h,t,d] buffers (q scaled by 1/sqrt(HD)).
// MODE 1: write f32 into outf[M,N].
// ---------------------------------------------------------------------------
template <int MODE>
__global__ __launch_bounds__(256) void gemm_bf16_kernel(
    const unsigned short* __restrict__ A,    // [M][K] bf16
    const unsigned short* __restrict__ Bw,   // [K][N] bf16
    const float* __restrict__ bias,          // [N]
    unsigned short* __restrict__ qb,
    unsigned short* __restrict__ kb,
    unsigned short* __restrict__ vb,
    float* __restrict__ outf,
    int N, int K) {
  __shared__ unsigned short As[64 * 32];     // [m][k]
  __shared__ unsigned short Bt[128 * 32];    // [n][k]  (transposed stage)

  const int m0 = blockIdx.y * 64;
  const int n0 = blockIdx.x * 128;
  const int tid = threadIdx.x;
  const int wave = tid >> 5, lane = tid & 31;
  const int wm = (wave >> 2) * 32;           // 0 / 32
  const int wn = (wave & 3) * 32;            // 0..96

  v8f acc[2][2] = {};

  const int arow = tid >> 2, acol = (tid & 3) * 8;   // A stage mapping
  const int bkk = tid >> 3, bng = (tid & 7) * 16;    // B stage mapping

  for (int k0 = 0; k0 < K; k0 += 32) {
    __syncthreads();
    // stage A: 64 rows x 32 k, async 16B per thread
    stage_b128(&A[(size_t)(m0 + arow) * K + k0 + acol], &As[arow * 32 + acol]);
    {   // stage B transposed: read [k][n] coalesced, write [n][k]
      const unsigned short* src = &Bw[(size_t)(k0 + bkk) * N + n0 + bng];
      FragU t;
      t.q[0] = *(const uint4*)(src);
      t.q[1] = *(const uint4*)(src + 8);
#pragma unroll
      for (int i = 0; i < 16; ++i) Bt[(bng + i) * 32 + bkk] = t.h[i];
    }
    if (k0 + 32 < K) {  // prefetch next B tile rows into L2/L0
      __builtin_prefetch(&Bw[(size_t)(k0 + 32 + bkk) * N + n0 + bng], 0, 1);
    }
    stage_wait();
    __syncthreads();

    v16bf a0 = ld_frag(&As[(wm + 0) * 32], 32, 0);
    v16bf a1 = ld_frag(&As[(wm + 16) * 32], 32, 0);
    v16bf b0 = ld_frag(&Bt[(wn + 0) * 32], 32, 0);
    v16bf b1 = ld_frag(&Bt[(wn + 16) * 32], 32, 0);
    acc[0][0] = wmma_bf16(a0, b0, acc[0][0]);
    acc[0][1] = wmma_bf16(a0, b1, acc[0][1]);
    acc[1][0] = wmma_bf16(a1, b0, acc[1][0]);
    acc[1][1] = wmma_bf16(a1, b1, acc[1][1]);
  }

  // Epilogue. C layout: lane holds column (lane&15), rows (lane>>4)*8 + r.
  const int col = lane & 15, rbase = (lane >> 4) * 8;
#pragma unroll
  for (int jm = 0; jm < 2; ++jm) {
#pragma unroll
    for (int jn = 0; jn < 2; ++jn) {
      int n = n0 + wn + jn * 16 + col;
      float bv = bias[n];
      if (MODE == 0) {
        int which = n >> 10;                 // 0=q 1=k 2=v (N tile aligns to 1024)
        int h = (n >> 6) & 15;
        int d = n & 63;
        unsigned short* dst = (which == 0) ? qb : (which == 1) ? kb : vb;
        float sc = (which == 0) ? 0.125f : 1.0f;   // 1/sqrt(64) folded into q
#pragma unroll
        for (int r = 0; r < 8; ++r) {
          int mrow = m0 + wm + jm * 16 + rbase + r;
          int b = mrow >> 11, t = mrow & (T_ - 1);
          float val = (acc[jm][jn][r] + bv) * sc;
          dst[((size_t)(b * NH_ + h) * T_ + t) * HD_ + d] = f2bf(val);
        }
      } else {
#pragma unroll
        for (int r = 0; r < 8; ++r) {
          int mrow = m0 + wm + jm * 16 + rbase + r;
          outf[(size_t)mrow * N + n] = acc[jm][jn][r] + bv;
        }
      }
    }
  }
}

// ---------------------------------------------------------------------------
// Flash attention (no causal mask, matching reference).
// grid = (T/64, B*NH). Block = 128 threads = 4 waves; wave owns 16 q-rows.
// ---------------------------------------------------------------------------
__global__ __launch_bounds__(128) void attn_kernel(
    const unsigned short* __restrict__ q,    // [b*NH][T][HD] bf16 (pre-scaled)
    const unsigned short* __restrict__ k,
    const unsigned short* __restrict__ v,
    unsigned short* __restrict__ y) {        // [M][H] bf16
  __shared__ unsigned short Kls[64 * 64];    // [key][d]   (= B^T layout for QK^T)
  __shared__ unsigned short Vt[64 * 64];     // [d][key]   (= B^T layout for PV)
  __shared__ unsigned short Pls[4 * 16 * 64];// per-wave [qrow][key] probs

  const int bh = blockIdx.y;
  const int q0 = blockIdx.x * 64;
  const int tid = threadIdx.x, wave = tid >> 5, lane = tid & 31;
  const int col = lane & 15, rbase = (lane >> 4) * 8;

  const unsigned short* qh = q + (size_t)bh * T_ * HD_;
  const unsigned short* kh = k + (size_t)bh * T_ * HD_;
  const unsigned short* vh = v + (size_t)bh * T_ * HD_;

  // Q strip (16 rows x 64 d), held in registers for the whole pass.
  const unsigned short* qs = qh + (size_t)(q0 + wave * 16) * HD_;
  v16bf aq0 = ld_frag(qs, HD_, 0);
  v16bf aq1 = ld_frag(qs, HD_, 32);

  v8f o[4] = {};
  float mrow[8], lrow[8];
#pragma unroll
  for (int r = 0; r < 8; ++r) { mrow[r] = -1e30f; lrow[r] = 0.0f; }

  const int srow = tid >> 1, shalf = (tid & 1) * 32;  // stage mapping

  for (int kbk = 0; kbk < T_ / 64; ++kbk) {
    __syncthreads();
    {   // cooperative stage: K row-major (async), V transposed (manual)
      const unsigned short* ksrc = kh + (size_t)(kbk * 64 + srow) * HD_ + shalf;
      stage_b128(ksrc + 0,  &Kls[srow * 64 + shalf + 0]);
      stage_b128(ksrc + 8,  &Kls[srow * 64 + shalf + 8]);
      stage_b128(ksrc + 16, &Kls[srow * 64 + shalf + 16]);
      stage_b128(ksrc + 24, &Kls[srow * 64 + shalf + 24]);

      const unsigned short* vsrc = vh + (size_t)(kbk * 64 + srow) * HD_ + shalf;
      union { uint4 u[4]; unsigned short h[32]; } tv;
      tv.u[0] = *(const uint4*)(vsrc + 0);
      tv.u[1] = *(const uint4*)(vsrc + 8);
      tv.u[2] = *(const uint4*)(vsrc + 16);
      tv.u[3] = *(const uint4*)(vsrc + 24);
#pragma unroll
      for (int i = 0; i < 32; ++i) Vt[(shalf + i) * 64 + srow] = tv.h[i];

      if (kbk + 1 < T_ / 64) {  // prefetch next K/V blocks
        const unsigned short* nks = kh + (size_t)((kbk + 1) * 64 + srow) * HD_ + shalf;
        const unsigned short* nvs = vh + (size_t)((kbk + 1) * 64 + srow) * HD_ + shalf;
        __builtin_prefetch(nks, 0, 1);
        __builtin_prefetch(nvs, 0, 1);
      }
    }
    stage_wait();
    __syncthreads();

    // S strip = Q(16x64) @ K^T(64x64): 4 column tiles x 2 K-steps.
    float s[4][8];
#pragma unroll
    for (int j = 0; j < 4; ++j) {
      v8f sa = {};
      v16bf bk0 = ld_frag(&Kls[(j * 16) * 64], 64, 0);
      v16bf bk1 = ld_frag(&Kls[(j * 16) * 64], 64, 32);
      sa = wmma_bf16(aq0, bk0, sa);
      sa = wmma_bf16(aq1, bk1, sa);
#pragma unroll
      for (int r = 0; r < 8; ++r) s[j][r] = sa[r];
    }

    // Online softmax; rows live across the 16 lanes of each half-wave.
#pragma unroll
    for (int r = 0; r < 8; ++r) {
      float mx = s[0][r];
#pragma unroll
      for (int j = 1; j < 4; ++j) mx = fmaxf(mx, s[j][r]);
#pragma unroll
      for (int off = 8; off >= 1; off >>= 1)
        mx = fmaxf(mx, __shfl_xor(mx, off, 32));
      float mn = fmaxf(mrow[r], mx);
      float sc = __expf(mrow[r] - mn);
      float rs = 0.0f;
#pragma unroll
      for (int j = 0; j < 4; ++j) {
        float p = __expf(s[j][r] - mn);
        s[j][r] = p;
        rs += p;
      }
#pragma unroll
      for (int off = 8; off >= 1; off >>= 1)
        rs += __shfl_xor(rs, off, 32);
      lrow[r] = lrow[r] * sc + rs;
      mrow[r] = mn;
#pragma unroll
      for (int jd = 0; jd < 4; ++jd) o[jd][r] *= sc;
#pragma unroll
      for (int j = 0; j < 4; ++j)
        Pls[(wave * 16 + rbase + r) * 64 + j * 16 + col] = f2bf(s[j][r]);
    }

    // O += P(16x64) @ V(64x64). P re-read from LDS in A-fragment layout
    // (same-wave LDS ops are in-order; compiler inserts the dscnt waits).
    v16bf ap0 = ld_frag(&Pls[wave * 16 * 64], 64, 0);
    v16bf ap1 = ld_frag(&Pls[wave * 16 * 64], 64, 32);
#pragma unroll
    for (int jd = 0; jd < 4; ++jd) {
      v16bf bv0 = ld_frag(&Vt[(jd * 16) * 64], 64, 0);
      v16bf bv1 = ld_frag(&Vt[(jd * 16) * 64], 64, 32);
      o[jd] = wmma_bf16(ap0, bv0, o[jd]);
      o[jd] = wmma_bf16(ap1, bv1, o[jd]);
    }
  }

  // Normalize and write y (bf16, [M][H] layout for the proj GEMM).
  const int b = bh >> 4, h = bh & 15;
#pragma unroll
  for (int jd = 0; jd < 4; ++jd) {
#pragma unroll
    for (int r = 0; r < 8; ++r) {
      int trow = q0 + wave * 16 + rbase + r;
      float val = o[jd][r] / lrow[r];
      y[(size_t)(b * T_ + trow) * H_ + h * HD_ + jd * 16 + col] = f2bf(val);
    }
  }
}

// ---------------------------------------------------------------------------
// Host-side launch
// ---------------------------------------------------------------------------
extern "C" void kernel_launch(void* const* d_in, const int* in_sizes, int n_in,
                              void* d_out, int out_size, void* d_ws, size_t ws_size,
                              hipStream_t stream) {
  const float* x  = (const float*)d_in[0];   // [B,T,H]
  const float* Wa = (const float*)d_in[1];   // [H,3H]
  const float* ba = (const float*)d_in[2];   // [3H]
  const float* Wp = (const float*)d_in[3];   // [H,H]
  const float* bp = (const float*)d_in[4];   // [H]
  float* out = (float*)d_out;                // [B,T,H] f32

  // Workspace carve-out (~48 MB), 256B aligned.
  char* w = (char*)d_ws;
  auto carve = [&](size_t bytes) {
    char* p = w;
    w += (bytes + 255) & ~(size_t)255;
    return (unsigned short*)p;
  };
  unsigned short* xb  = carve((size_t)M_ * H_ * 2);       // x bf16
  unsigned short* wab = carve((size_t)H_ * N_QKV * 2);    // W_attn bf16
  unsigned short* wpb = carve((size_t)H_ * H_ * 2);       // W_proj bf16
  unsigned short* qb  = carve((size_t)M_ * H_ * 2);       // q [b,h,t,d]
  unsigned short* kb  = carve((size_t)M_ * H_ * 2);       // k [b,h,t,d]
  unsigned short* vb  = carve((size_t)M_ * H_ * 2);       // v [b,h,t,d]
  unsigned short* yb  = carve((size_t)M_ * H_ * 2);       // attn out [M,H]

  // 1) f32 -> bf16 conversions
  cvt_f32_bf16<<<2048, 256, 0, stream>>>(x,  xb,  M_ * H_);
  cvt_f32_bf16<<<2048, 256, 0, stream>>>(Wa, wab, H_ * N_QKV);
  cvt_f32_bf16<<<1024, 256, 0, stream>>>(Wp, wpb, H_ * H_);

  // 2) QKV GEMM + bias + head-major scatter (q pre-scaled by 1/sqrt(HD))
  gemm_bf16_kernel<0><<<dim3(N_QKV / 128, M_ / 64), 256, 0, stream>>>(
      xb, wab, ba, qb, kb, vb, nullptr, N_QKV, H_);

  // 3) Fused flash attention (softmax, no mask)
  attn_kernel<<<dim3(T_ / 64, B_ * NH_), 128, 0, stream>>>(qb, kb, vb, yb);

  // 4) Output projection + bias -> f32
  gemm_bf16_kernel<1><<<dim3(H_ / 128, M_ / 64), 256, 0, stream>>>(
      yb, wpb, bp, nullptr, nullptr, nullptr, out, H_, H_);
}
